// MultiHeadAttention_79611513799147
// MI455X (gfx1250) — matrix operand order repaired and checked by reading, not compiled, not run on hardware
//
#include <hip/hip_runtime.h>
#include <hip/hip_bf16.h>
#include <math.h>

// ---------------------------------------------------------------------------
// Problem constants (hardcoded from the reference)
// ---------------------------------------------------------------------------
#define D_MODEL 1024
#define N_HEADS 16
#define D_HEAD  64
#define BATCH   2
#define LQ      2048
#define LK      2048
#define M_ROWS  (BATCH * LQ)      // 4096
#define NUM_BUCKETS 32
#define BIAS_TAB 4096             // per-head bias table length (d + 2047)

typedef __attribute__((ext_vector_type(16))) __bf16 v16bf;
typedef __attribute__((ext_vector_type(8)))  float  v8f;

union Frag { uint4 u[2]; v16bf v; };

// ---- optional CDNA5 async global->LDS path (guarded; falls back cleanly) ---
#if defined(__gfx1250__) && defined(__has_builtin)
#if __has_builtin(__builtin_amdgcn_global_load_async_to_lds_b128) && \
    __has_builtin(__builtin_amdgcn_s_wait_asynccnt)
#define HAVE_ASYNC_LDS 1
#endif
#endif

#if defined(HAVE_ASYNC_LDS)
// Signature learned from the round-2 diagnostic: param type is
// "int __attribute__((__vector_size__(16))) *" (generic pointer), 4 args.
typedef int v4i_vs __attribute__((vector_size(16)));
static __device__ __forceinline__ void asyncCopy16(const void* g, void* l) {
    // global_load_async_to_lds_b128: 16 bytes per lane, tracked by ASYNCcnt
    __builtin_amdgcn_global_load_async_to_lds_b128(
        (v4i_vs*)(size_t)g, (v4i_vs*)(size_t)l, 0, 0);
}
static __device__ __forceinline__ void asyncWait() {
    __builtin_amdgcn_s_wait_asynccnt(0);
}
#endif

static __device__ __forceinline__ __bf16 f2bf(float f) {
    unsigned u = __builtin_bit_cast(unsigned, f);
    unsigned r = u + 0x7FFFu + ((u >> 16) & 1u);   // round-to-nearest-even
    unsigned short s = (unsigned short)(r >> 16);
    return __builtin_bit_cast(__bf16, s);
}

// A-matrix fragment (16x32 bf16, MxK), row-major LDS tile with leading dim ldk.
// Layout (ISA 7.12.2): lanes 0-15 hold K=0..7,16..23 ; lanes 16-31 hold K=8..15,24..31
static __device__ __forceinline__ v16bf ldsFragA(const __bf16* base, int row,
                                                 int ldk, int kbase, int lane) {
    const int kg = (lane >> 4) & 1;
    const __bf16* p = base + row * ldk + kbase + kg * 8;
    Frag f;
    f.u[0] = *reinterpret_cast<const uint4*>(p);
    f.u[1] = *reinterpret_cast<const uint4*>(p + 16);
    return f.v;
}

// B-matrix fragment (32x16 bf16, KxN) where LDS tile stores B transposed as
// [n][k] row-major (same K order as A): lanes 0-15 K=0..7,16..23 ; 16-31 K=8..15,24..31
static __device__ __forceinline__ v16bf ldsFragB(const __bf16* base, int col,
                                                 int ldk, int kbase, int lane) {
    const int kh = (lane >> 4) & 1;
    const __bf16* p = base + col * ldk + kbase + kh * 8;
    const uint4* q = reinterpret_cast<const uint4*>(p);
    Frag f;
    f.u[0] = q[0];
    f.u[1] = q[2];
    return f.v;
}

static __device__ __forceinline__ v8f wmma_bf16(v16bf a, v16bf b, v8f c) {
    return __builtin_amdgcn_wmma_f32_16x16x32_bf16(false, a, false, b,
                                                   (short)0, c, false, false);
}

// ---------------------------------------------------------------------------
// Elementwise fp32 -> bf16
// ---------------------------------------------------------------------------
__global__ __launch_bounds__(256)
void cvt_f32_bf16_kernel(const float* __restrict__ src, __bf16* __restrict__ dst, int n) {
    int i = (blockIdx.x * blockDim.x + threadIdx.x) * 4;
    if (i + 3 < n) {
        float4 f = *reinterpret_cast<const float4*>(src + i);
        union { unsigned long long u; __bf16 e[4]; } o;
        o.e[0] = f2bf(f.x); o.e[1] = f2bf(f.y);
        o.e[2] = f2bf(f.z); o.e[3] = f2bf(f.w);
        *reinterpret_cast<unsigned long long*>(dst + i) = o.u;
    }
}

// ---------------------------------------------------------------------------
// Precompute relative-position bias table: biasT[h][d + 2047] = rel_emb[bucket(d)][h]
// ---------------------------------------------------------------------------
__global__ __launch_bounds__(256)
void bias_table_kernel(const float* __restrict__ rel_emb, float* __restrict__ biasT) {
    int idx = blockIdx.x * blockDim.x + threadIdx.x;
    if (idx >= N_HEADS * BIAS_TAB) return;
    int h = idx >> 12;
    int i = idx & (BIAS_TAB - 1);
    int d = i - 2047;                 // d = k - q  (rel = mem - ctx)
    int n = d < 0 ? -d : 0;           // n = max(-rel, 0)
    int bucket;
    if (n < 16) {
        bucket = n;
    } else {
        float t = logf((float)n / 16.0f + 1e-9f) / logf(8.0f) * 16.0f;
        int vl = 16 + (int)t;         // trunc toward zero like astype(int32)
        bucket = vl < (NUM_BUCKETS - 1) ? vl : (NUM_BUCKETS - 1);
    }
    biasT[idx] = rel_emb[bucket * N_HEADS + h];
}

// ---------------------------------------------------------------------------
// Tiled WMMA GEMM:  C[M=4096, N=1024] = A[4096,1024](bf16) @ W[1024,1024]^T(bf16)
//   mode 0: output bf16 head-split [B,H,L,dh]
//   mode 1: output fp32 row-major (d_out)
// Block: 128x128 C-tile, K-step 32, 8 waves (wave tile 32x64 -> 2x4 WMMA tiles)
// Double-buffered LDS; async global->LDS when the toolchain exposes it.
// ---------------------------------------------------------------------------
__global__ __launch_bounds__(256)
void gemm_wmma_kernel(const __bf16* __restrict__ A,
                      const __bf16* __restrict__ Wb,
                      __bf16* __restrict__ outBf,
                      float* __restrict__ outF32,
                      int mode) {
    __shared__ __align__(16) __bf16 As[2][128 * 32];
    __shared__ __align__(16) __bf16 Bs[2][128 * 32];

    const int tid  = threadIdx.x;
    const int lane = tid & 31;
    const int wave = tid >> 5;
    const int wm   = wave >> 1;           // 0..3
    const int wn   = wave & 1;            // 0..1
    const int half = (lane >> 4) & 1;
    const int nl   = lane & 15;
    const int n0   = blockIdx.x * 128;
    const int m0   = blockIdx.y * 128;
    const int lrow = tid >> 1;            // 0..127
    const int lseg = (tid & 1) * 16;      // 0 or 16

    const __bf16* aSrc = A  + (size_t)(m0 + lrow) * D_MODEL + lseg;
    const __bf16* wSrc = Wb + (size_t)(n0 + lrow) * D_MODEL + lseg;
    __bf16* aDst = &As[0][lrow * 32 + lseg];
    __bf16* wDst = &Bs[0][lrow * 32 + lseg];
    const int bufStride = 128 * 32;

    v8f acc[2][4];
#pragma unroll
    for (int i = 0; i < 2; ++i)
#pragma unroll
        for (int j = 0; j < 4; ++j)
#pragma unroll
            for (int e = 0; e < 8; ++e) acc[i][j][e] = 0.f;

    constexpr int NT = D_MODEL / 32;      // 32 K-steps

    // ---- stage tile 0 into buffer 0 ----
#if defined(HAVE_ASYNC_LDS)
    asyncCopy16(aSrc,     aDst);
    asyncCopy16(aSrc + 8, aDst + 8);
    asyncCopy16(wSrc,     wDst);
    asyncCopy16(wSrc + 8, wDst + 8);
#else
    {
        const uint4* ag = reinterpret_cast<const uint4*>(aSrc);
        const uint4* wg = reinterpret_cast<const uint4*>(wSrc);
        uint4* ad = reinterpret_cast<uint4*>(aDst);
        uint4* wd = reinterpret_cast<uint4*>(wDst);
        ad[0] = ag[0]; ad[1] = ag[1];
        wd[0] = wg[0]; wd[1] = wg[1];
    }
#endif

    for (int kt = 0; kt < NT; ++kt) {
        const int cur = kt & 1;
        const int nxt = cur ^ 1;
        const int kn  = (kt + 1) * 32;    // element offset of next tile

#if defined(HAVE_ASYNC_LDS)
        asyncWait();                      // our async stores to LDS landed
        __syncthreads();                  // everyone's staging visible
        if (kt + 1 < NT) {                // fire next tile, overlap with WMMA
            asyncCopy16(aSrc + kn,     aDst + nxt * bufStride);
            asyncCopy16(aSrc + kn + 8, aDst + nxt * bufStride + 8);
            asyncCopy16(wSrc + kn,     wDst + nxt * bufStride);
            asyncCopy16(wSrc + kn + 8, wDst + nxt * bufStride + 8);
        }
#else
        __syncthreads();
        Frag ra, rw;
        if (kt + 1 < NT) {                // issue loads; land them after WMMAs
            const uint4* ag = reinterpret_cast<const uint4*>(aSrc + kn);
            const uint4* wg = reinterpret_cast<const uint4*>(wSrc + kn);
            ra.u[0] = ag[0]; ra.u[1] = ag[1];
            rw.u[0] = wg[0]; rw.u[1] = wg[1];
        }
#endif

        // ---- compute on current buffer ----
        const __bf16* aBuf = &As[cur][0];
        const __bf16* bBuf = &Bs[cur][0];
        v16bf afrag[2], bfrag[4];
#pragma unroll
        for (int i = 0; i < 2; ++i)
            afrag[i] = ldsFragA(aBuf, wm * 32 + i * 16 + nl, 32, 0, lane);
#pragma unroll
        for (int j = 0; j < 4; ++j)
            bfrag[j] = ldsFragB(bBuf, wn * 64 + j * 16 + nl, 32, 0, lane);
#pragma unroll
        for (int i = 0; i < 2; ++i)
#pragma unroll
            for (int j = 0; j < 4; ++j)
                acc[i][j] = wmma_bf16(afrag[i], bfrag[j], acc[i][j]);

#if !defined(HAVE_ASYNC_LDS)
        if (kt + 1 < NT) {
            uint4* ad = reinterpret_cast<uint4*>(aDst + nxt * bufStride);
            uint4* wd = reinterpret_cast<uint4*>(wDst + nxt * bufStride);
            ad[0] = ra.u[0]; ad[1] = ra.u[1];
            wd[0] = rw.u[0]; wd[1] = rw.u[1];
        }
#endif
    }

    // ---- epilogue: C layout = lane holds N=lane&15, rows v + 8*(lane>>4) ----
#pragma unroll
    for (int i = 0; i < 2; ++i) {
#pragma unroll
        for (int j = 0; j < 4; ++j) {
#pragma unroll
            for (int v = 0; v < 8; ++v) {
                int m = m0 + wm * 32 + i * 16 + v + 8 * half;
                int n = n0 + wn * 64 + j * 16 + nl;
                float val = acc[i][j][v];
                if (mode == 0) {
                    int b = m >> 11, l = m & (LQ - 1);
                    int h = n >> 6, dh = n & (D_HEAD - 1);
                    outBf[(((size_t)(b * N_HEADS + h) * LQ) + l) * D_HEAD + dh] = f2bf(val);
                } else {
                    outF32[(size_t)m * D_MODEL + n] = val;
                }
            }
        }
    }
}

// ---------------------------------------------------------------------------
// Flash attention: block = 64 q-rows of one (b,h); 4 waves x 16 rows.
// KV loop in 64-column tiles; S and P*V via bf16 WMMA; fp32 online softmax.
// ---------------------------------------------------------------------------
__global__ __launch_bounds__(128)
void flash_attn_kernel(const __bf16* __restrict__ qh,
                       const __bf16* __restrict__ kh,
                       const __bf16* __restrict__ vh,
                       const unsigned char* __restrict__ mask,
                       const float* __restrict__ biasT,
                       __bf16* __restrict__ Obf) {
    __shared__ __align__(16) __bf16 Qs[64 * 64];
    __shared__ __align__(16) __bf16 Ks[64 * 64];
    __shared__ __align__(16) __bf16 Vt[64 * 64];        // transposed [dh][kv]
    __shared__ __align__(16) __bf16 Ps[4][16 * 64];     // per-wave P tile
    __shared__ float biasLds[128];
    __shared__ float maskLds[64];

    const int tid  = threadIdx.x;
    const int lane = tid & 31;
    const int w    = tid >> 5;
    const int half = (lane >> 4) & 1;
    const int nl   = lane & 15;
    const int q0   = blockIdx.x * 64;
    const int bh   = blockIdx.y;
    const int b    = bh >> 4;
    const int h    = bh & (N_HEADS - 1);
    const int lrow = tid >> 1;            // 0..63
    const int lseg = (tid & 1) * 32;      // 0 or 32

    // load Q tile (64 rows x 64 dh bf16)
    {
        const uint4* src = reinterpret_cast<const uint4*>(
            qh + ((size_t)bh * LQ + q0 + lrow) * D_HEAD + lseg);
        uint4* dst = reinterpret_cast<uint4*>(&Qs[lrow * 64 + lseg]);
        dst[0] = src[0]; dst[1] = src[1]; dst[2] = src[2]; dst[3] = src[3];
    }

    float mrow[8], lsum[8];
    v8f o[4];
#pragma unroll
    for (int v = 0; v < 8; ++v) { mrow[v] = -3.0e38f; lsum[v] = 0.f; }
#pragma unroll
    for (int j = 0; j < 4; ++j)
#pragma unroll
        for (int e = 0; e < 8; ++e) o[j][e] = 0.f;

    for (int kv0 = 0; kv0 < LK; kv0 += 64) {
        // ---- stage K tile + transposed V tile ----
        {
            const uint4* ksrc = reinterpret_cast<const uint4*>(
                kh + ((size_t)bh * LK + kv0 + lrow) * D_HEAD + lseg);
            uint4* kdst = reinterpret_cast<uint4*>(&Ks[lrow * 64 + lseg]);
            kdst[0] = ksrc[0]; kdst[1] = ksrc[1]; kdst[2] = ksrc[2]; kdst[3] = ksrc[3];

            union { uint4 u[4]; __bf16 e[32]; } vv;
            const uint4* vsrc = reinterpret_cast<const uint4*>(
                vh + ((size_t)bh * LK + kv0 + lrow) * D_HEAD + lseg);
            vv.u[0] = vsrc[0]; vv.u[1] = vsrc[1]; vv.u[2] = vsrc[2]; vv.u[3] = vsrc[3];
#pragma unroll
            for (int e = 0; e < 32; ++e) Vt[(lseg + e) * 64 + lrow] = vv.e[e];
        }
        if (tid < 64)
            maskLds[tid] = mask[(size_t)b * LK + kv0 + tid] ? 0.f : 1.f;
        {   // bias strip: index i -> d = (kv0 - q0) + (i - 64)
            int tix = (kv0 - q0) + (tid - 64) + 2047;
            tix = tix < 0 ? 0 : (tix > (BIAS_TAB - 2) ? (BIAS_TAB - 2) : tix);
            biasLds[tid] = biasT[h * BIAS_TAB + tix];
        }
        if (kv0 + 64 < LK) {   // prefetch next K/V rows while we compute
            __builtin_prefetch(kh + ((size_t)bh * LK + kv0 + 64 + lrow) * D_HEAD + lseg, 0, 1);
            __builtin_prefetch(vh + ((size_t)bh * LK + kv0 + 64 + lrow) * D_HEAD + lseg, 0, 1);
        }
        __syncthreads();

        // ---- S = Q K^T  (wave handles rows [16w,16w+16) x 64 kv cols) ----
        v8f s[4];
#pragma unroll
        for (int j = 0; j < 4; ++j)
#pragma unroll
            for (int e = 0; e < 8; ++e) s[j][e] = 0.f;
#pragma unroll
        for (int kk = 0; kk < 2; ++kk) {
            v16bf a = ldsFragA(Qs, w * 16 + nl, 64, kk * 32, lane);
#pragma unroll
            for (int j = 0; j < 4; ++j) {
                v16bf bb = ldsFragB(Ks, j * 16 + nl, 64, kk * 32, lane);
                s[j] = wmma_bf16(a, bb, s[j]);
            }
        }

        // ---- scale + rel-pos bias + key padding mask ----
#pragma unroll
        for (int j = 0; j < 4; ++j) {
            int c = j * 16 + nl;
            float masked = maskLds[c];
#pragma unroll
            for (int v = 0; v < 8; ++v) {
                int rblk = w * 16 + v + 8 * half;
                float val = s[j][v] * 0.125f + biasLds[c - rblk + 64];
                s[j][v] = (masked != 0.f) ? -1.0e9f : val;
            }
        }

        // ---- online softmax (row reductions across 16-lane half-groups) ----
        float fac[8];
#pragma unroll
        for (int v = 0; v < 8; ++v) {
            float mx = s[0][v];
#pragma unroll
            for (int j = 1; j < 4; ++j) mx = fmaxf(mx, s[j][v]);
            mx = fmaxf(mx, __shfl_xor(mx, 1, 32));
            mx = fmaxf(mx, __shfl_xor(mx, 2, 32));
            mx = fmaxf(mx, __shfl_xor(mx, 4, 32));
            mx = fmaxf(mx, __shfl_xor(mx, 8, 32));
            float mnew = fmaxf(mrow[v], mx);
            fac[v]  = __expf(mrow[v] - mnew);
            mrow[v] = mnew;
            float rs = 0.f;
#pragma unroll
            for (int j = 0; j < 4; ++j) {
                float p = __expf(s[j][v] - mnew);
                s[j][v] = p;
                rs += p;
            }
            rs += __shfl_xor(rs, 1, 32);
            rs += __shfl_xor(rs, 2, 32);
            rs += __shfl_xor(rs, 4, 32);
            rs += __shfl_xor(rs, 8, 32);
            lsum[v] = lsum[v] * fac[v] + rs;
        }

        // ---- spill P (bf16) to per-wave LDS; rescale O accumulators ----
#pragma unroll
        for (int j = 0; j < 4; ++j) {
            int c = j * 16 + nl;
#pragma unroll
            for (int v = 0; v < 8; ++v) {
                Ps[w][(v + 8 * half) * 64 + c] = f2bf(s[j][v]);
                o[j][v] *= fac[v];
            }
        }
        asm volatile("s_wait_dscnt 0" ::: "memory");   // in-wave store->load

        // ---- O += P @ V  (Vt gives contiguous B fragments) ----
#pragma unroll
        for (int kk = 0; kk < 2; ++kk) {
            v16bf a = ldsFragA(&Ps[w][0], nl, 64, kk * 32, lane);
#pragma unroll
            for (int j = 0; j < 4; ++j) {
                v16bf bb = ldsFragB(Vt, j * 16 + nl, 64, kk * 32, lane);
                o[j] = wmma_bf16(a, bb, o[j]);
            }
        }
        __syncthreads();
    }

    // ---- write O (bf16, merged [B, L, D] layout for the Wo GEMM) ----
#pragma unroll
    for (int j = 0; j < 4; ++j) {
#pragma unroll
        for (int v = 0; v < 8; ++v) {
            int row = q0 + w * 16 + v + 8 * half;
            int col = h * D_HEAD + j * 16 + nl;
            float val = o[j][v] / lsum[v];
            Obf[((size_t)b * LQ + row) * D_MODEL + col] = f2bf(val);
        }
    }
}

// ---------------------------------------------------------------------------
// Launch
// ---------------------------------------------------------------------------
extern "C" void kernel_launch(void* const* d_in, const int* in_sizes, int n_in,
                              void* d_out, int out_size, void* d_ws, size_t ws_size,
                              hipStream_t stream) {
    const float* q  = (const float*)d_in[0];
    const float* k  = (const float*)d_in[1];
    const float* v  = (const float*)d_in[2];
    const unsigned char* mask = (const unsigned char*)d_in[3];
    const float* Wq = (const float*)d_in[4];
    const float* Wk = (const float*)d_in[5];
    const float* Wv = (const float*)d_in[6];
    const float* Wo = (const float*)d_in[7];
    const float* rel_emb = (const float*)d_in[8];

    char* ws = (char*)d_ws;
    const size_t WBYTES = (size_t)D_MODEL * D_MODEL * sizeof(__bf16);   // 2 MB
    const size_t XBYTES = (size_t)M_ROWS * D_MODEL * sizeof(__bf16);    // 8 MB
    size_t off = 0;
    __bf16* Wqb = (__bf16*)(ws + off); off += WBYTES;
    __bf16* Wkb = (__bf16*)(ws + off); off += WBYTES;
    __bf16* Wvb = (__bf16*)(ws + off); off += WBYTES;
    __bf16* Wob = (__bf16*)(ws + off); off += WBYTES;
    __bf16* qB  = (__bf16*)(ws + off); off += XBYTES;   // bf16 copies of inputs
    __bf16* kB  = (__bf16*)(ws + off); off += XBYTES;
    __bf16* vB  = (__bf16*)(ws + off); off += XBYTES;
    __bf16* qhB = (__bf16*)(ws + off); off += XBYTES;   // head-split projections
    __bf16* khB = (__bf16*)(ws + off); off += XBYTES;
    __bf16* vhB = (__bf16*)(ws + off); off += XBYTES;
    __bf16* ObF = (__bf16*)(ws + off); off += XBYTES;   // attention output
    float* biasT = (float*)(ws + off);                  // 256 KB

    const int WN = D_MODEL * D_MODEL;   // 1M elems per weight
    const int XN = M_ROWS * D_MODEL;    // 4M elems per activation
    cvt_f32_bf16_kernel<<<WN / 1024, 256, 0, stream>>>(Wq, Wqb, WN);
    cvt_f32_bf16_kernel<<<WN / 1024, 256, 0, stream>>>(Wk, Wkb, WN);
    cvt_f32_bf16_kernel<<<WN / 1024, 256, 0, stream>>>(Wv, Wvb, WN);
    cvt_f32_bf16_kernel<<<WN / 1024, 256, 0, stream>>>(Wo, Wob, WN);
    cvt_f32_bf16_kernel<<<XN / 1024, 256, 0, stream>>>(q, qB, XN);
    cvt_f32_bf16_kernel<<<XN / 1024, 256, 0, stream>>>(k, kB, XN);
    cvt_f32_bf16_kernel<<<XN / 1024, 256, 0, stream>>>(v, vB, XN);

    bias_table_kernel<<<(N_HEADS * BIAS_TAB) / 256, 256, 0, stream>>>(rel_emb, biasT);

    dim3 gemmGrid(D_MODEL / 128, M_ROWS / 128);   // (8, 32)
    gemm_wmma_kernel<<<gemmGrid, 256, 0, stream>>>(qB, Wqb, qhB, nullptr, 0);
    gemm_wmma_kernel<<<gemmGrid, 256, 0, stream>>>(kB, Wkb, khB, nullptr, 0);
    gemm_wmma_kernel<<<gemmGrid, 256, 0, stream>>>(vB, Wvb, vhB, nullptr, 0);

    dim3 faGrid(LQ / 64, BATCH * N_HEADS);        // (32, 32)
    flash_attn_kernel<<<faGrid, 128, 0, stream>>>(qhB, khB, vhB, mask, biasT, ObF);

    gemm_wmma_kernel<<<gemmGrid, 256, 0, stream>>>(ObF, Wob, nullptr, (float*)d_out, 1);
}
